// PairwiseProjHead_49220325212325
// MI455X (gfx1250) — matrix operand order repaired and checked by reading, not compiled, not run on hardware
//
#include <hip/hip_runtime.h>
#include <hip/hip_bf16.h>

typedef __attribute__((ext_vector_type(16))) _Float16 v16h;
typedef __attribute__((ext_vector_type(8)))  _Float16 v8h;
typedef __attribute__((ext_vector_type(8)))  float    v8f;

#define DM   512   // D_MODEL
#define NSEQ 512
#define NB   2
#define DHID 100   // 64 + 36
#define NFRAG (28*32*16)  // 4 ktiles * 7 ntiles * 32 lanes * 16 halves

static __device__ __forceinline__ v16h cat8(v8h lo, v8h hi) {
    return __builtin_shufflevector(lo, hi, 0,1,2,3,4,5,6,7,8,9,10,11,12,13,14,15);
}

static __device__ __forceinline__ float fast_silu(float a) {
    // a * sigmoid(a), using v_exp_f32 + v_rcp_f32 (no IEEE div sequence)
    float e = __builtin_amdgcn_exp2f(a * -1.44269504088896340736f); // exp(-a)
    return a * __builtin_amdgcn_rcpf(1.0f + e);
}

// ---------------------------------------------------------------------------
// Kernel 1: qk = x @ W_down + b_down ; split into q,k stored as f16 rows of 64
// ---------------------------------------------------------------------------
__global__ __launch_bounds__(128)
void proj_kernel(const float* __restrict__ x, const float* __restrict__ Wd,
                 const float* __restrict__ bd,
                 _Float16* __restrict__ qws, _Float16* __restrict__ kws)
{
    __shared__ float xrow[DM];
    const int row = blockIdx.x;                 // b*512 + n
    const float* xr = x + (size_t)row * DM;
    for (int d = threadIdx.x; d < DM; d += 128) xrow[d] = xr[d];
    __syncthreads();
    const int c = threadIdx.x;                  // 0..127
    float acc = bd[c];
    #pragma unroll 8
    for (int d = 0; d < DM; ++d) acc += xrow[d] * Wd[d * 128 + c];
    if (c < 64) qws[row * 64 + c]        = (_Float16)acc;
    else        kws[row * 64 + (c - 64)] = (_Float16)acc;
}

// ---------------------------------------------------------------------------
// Kernel 2: pre-pack W1 (128x100) and W2 (100x100, K-padded to 128) into the
// per-lane WMMA B-fragment layout: B 32x16 f16, lane L: N = L%16,
// K = 32*kt + (L/16)*8 + (h<8 ? h : 8+h)  (h = half index 0..15). Padded entries are 0.
// ---------------------------------------------------------------------------
__global__ __launch_bounds__(256)
void pack_w_kernel(const float* __restrict__ W1, const float* __restrict__ W2,
                   _Float16* __restrict__ w1f, _Float16* __restrict__ w2f)
{
    int idx = blockIdx.x * 256 + threadIdx.x;
    if (idx >= NFRAG) return;
    int f    = idx >> 9;          // fragment index 0..27
    int rem  = idx & 511;
    int lane = rem >> 4;
    int h    = rem & 15;
    int kt = f / 7, nt = f % 7;
    int K = 32 * kt + (lane >> 4) * 8 + (h < 8 ? h : 8 + h);
    int N = nt * 16 + (lane & 15);
    float v1 = (N < DHID)             ? W1[K * DHID + N] : 0.0f;  // K < 128 always
    float v2 = (N < DHID && K < DHID) ? W2[K * DHID + N] : 0.0f;
    w1f[idx] = (_Float16)v1;
    w2f[idx] = (_Float16)v2;
}

// ---------------------------------------------------------------------------
// Kernel 3: pairwise MLP. One block per (b,i); wave w does j-tiles
// [ (4w+t)*16 , +16 ) for t=0..3. Per tile: 56 v_wmma_f32_16x16x32_f16.
// nt loops NOT unrolled (bounds live B fragments); B fragments loaded into
// distinct registers so all 8 b128 loads clause together before the WMMAs.
// ---------------------------------------------------------------------------
__global__ __launch_bounds__(256)
void pair_mlp_kernel(const _Float16* __restrict__ qws,
                     const _Float16* __restrict__ kws,
                     const v16h* __restrict__ w1f, const v16h* __restrict__ w2f,
                     const float* __restrict__ b1, const float* __restrict__ b2,
                     float* __restrict__ out)
{
    // per-wave h scratch: 16 rows x 144 halves (pad to kill bank conflicts)
    __shared__ _Float16 hbuf[8 * 16 * 144];

    const int bi   = blockIdx.x;          // b*512 + i
    const int b    = bi >> 9;
    const int lane = threadIdx.x & 31;
    const int wave = threadIdx.x >> 5;
    const int lh   = lane >> 4;           // lane half (0/1)
    const int ln   = lane & 15;
    const int c    = lh * 8;              // per-lane K sub-offset

    _Float16* hrow = hbuf + wave * (16 * 144);

    // q_i parts (uniform i per block); A-layout needs runs [c..c+7],[16+c..23+c],...
    const _Float16* qr = qws + (size_t)bi * 64;
    const v8h q0a = *(const v8h*)(qr + c);
    const v8h q0b = *(const v8h*)(qr + 16 + c);
    const v8h q1a = *(const v8h*)(qr + 32 + c);
    const v8h q1b = *(const v8h*)(qr + 48 + c);

    const size_t anglBase = (size_t)NB * NSEQ * NSEQ * 64;

    #pragma unroll 1
    for (int t = 0; t < 4; ++t) {
        const int j0 = (wave * 4 + t) * 16;
        const int j  = j0 + ln;                     // this lane's pair row (M = ln)
        const _Float16* kr = kws + ((size_t)(b * NSEQ + j)) * 64;
        v8h k0a = *(const v8h*)(kr + c);
        v8h k0b = *(const v8h*)(kr + 16 + c);
        v8h k1a = *(const v8h*)(kr + 32 + c);
        v8h k1b = *(const v8h*)(kr + 48 + c);

        // pairwise features: [0..63] = q*k ; [64..127] = k - q
        v16h A0 = cat8(q0a * k0a, q0b * k0b);
        v16h A1 = cat8(q1a * k1a, q1b * k1b);
        v16h A2 = cat8(k0a - q0a, k0b - q0b);
        v16h A3 = cat8(k1a - q1a, k1b - q1b);

        // ---------- layer 1: (16x128) @ (128x112) + b1, SiLU -> LDS ----------
        #pragma unroll 1
        for (int nt = 0; nt < 7; ++nt) {
            int o = nt * 16 + ln;
            float bias = (o < DHID) ? b1[o] : 0.0f;
            const v16h* wp = w1f + nt * 32 + lane;
            v16h B0 = wp[0 * 7 * 32];
            v16h B1 = wp[1 * 7 * 32];
            v16h B2 = wp[2 * 7 * 32];
            v16h B3 = wp[3 * 7 * 32];
            v8f acc;
            #pragma unroll
            for (int e = 0; e < 8; ++e) acc[e] = bias;
            acc = __builtin_amdgcn_wmma_f32_16x16x32_f16(false, A0, false, B0,
                                                         (short)0, acc, false, false);
            acc = __builtin_amdgcn_wmma_f32_16x16x32_f16(false, A1, false, B1,
                                                         (short)0, acc, false, false);
            acc = __builtin_amdgcn_wmma_f32_16x16x32_f16(false, A2, false, B2,
                                                         (short)0, acc, false, false);
            acc = __builtin_amdgcn_wmma_f32_16x16x32_f16(false, A3, false, B3,
                                                         (short)0, acc, false, false);
            #pragma unroll
            for (int v = 0; v < 8; ++v)
                hrow[(v + 8 * lh) * 144 + o] = (_Float16)fast_silu(acc[v]);
        }
        // zero-pad hidden cols 112..127 (cols 100..111 already silu(0)=0)
        {
            v8h z = {};
            *(v8h*)(hrow + ln * 144 + 112 + c) = z;
        }

        // ---------- layer 2 A fragments from LDS (transpose via layout) ------
        v16h Ah0, Ah1, Ah2, Ah3;
        {
            const _Float16* hr = hrow + ln * 144 + c;
            Ah0 = cat8(*(const v8h*)(hr +  0), *(const v8h*)(hr + 16));
            Ah1 = cat8(*(const v8h*)(hr + 32), *(const v8h*)(hr + 48));
            Ah2 = cat8(*(const v8h*)(hr + 64), *(const v8h*)(hr + 80));
            Ah3 = cat8(*(const v8h*)(hr + 96), *(const v8h*)(hr + 112));
        }

        // ---------- layer 2: (16x128) @ (128x112) + b2 -> global -------------
        #pragma unroll 1
        for (int nt = 0; nt < 7; ++nt) {
            int o = nt * 16 + ln;                       // output channel
            float bias = (o < DHID) ? b2[o] : 0.0f;
            const v16h* wp = w2f + nt * 32 + lane;
            v16h B0 = wp[0 * 7 * 32];
            v16h B1 = wp[1 * 7 * 32];
            v16h B2 = wp[2 * 7 * 32];
            v16h B3 = wp[3 * 7 * 32];
            v8f acc;
            #pragma unroll
            for (int e = 0; e < 8; ++e) acc[e] = bias;
            acc = __builtin_amdgcn_wmma_f32_16x16x32_f16(false, Ah0, false, B0,
                                                         (short)0, acc, false, false);
            acc = __builtin_amdgcn_wmma_f32_16x16x32_f16(false, Ah1, false, B1,
                                                         (short)0, acc, false, false);
            acc = __builtin_amdgcn_wmma_f32_16x16x32_f16(false, Ah2, false, B2,
                                                         (short)0, acc, false, false);
            acc = __builtin_amdgcn_wmma_f32_16x16x32_f16(false, Ah3, false, B3,
                                                         (short)0, acc, false, false);
            #pragma unroll
            for (int v = 0; v < 8; ++v) {
                int m = v + 8 * lh;                     // local pair row
                size_t pidx = (size_t)bi * NSEQ + (j0 + m);
                if (o < 64)
                    out[pidx * 64 + o] = acc[v];                       // distogram
                else if (o < DHID)
                    out[anglBase + pidx * 36 + (o - 64)] = acc[v];     // anglogram
            }
        }
    }
}

// ---------------------------------------------------------------------------
extern "C" void kernel_launch(void* const* d_in, const int* in_sizes, int n_in,
                              void* d_out, int out_size, void* d_ws, size_t ws_size,
                              hipStream_t stream)
{
    const float* x  = (const float*)d_in[0];
    const float* Wd = (const float*)d_in[1];
    const float* bd = (const float*)d_in[2];
    const float* W1 = (const float*)d_in[3];
    const float* b1 = (const float*)d_in[4];
    const float* W2 = (const float*)d_in[5];
    const float* b2 = (const float*)d_in[6];
    float* out = (float*)d_out;

    // workspace layout (needs ~313 KB):
    //   q  f16 [2*512*64] : 131072 B
    //   k  f16 [2*512*64] : 131072 B
    //   W1 fragments      :  28672 B
    //   W2 fragments      :  28672 B
    char* ws = (char*)d_ws;
    _Float16* qws = (_Float16*)(ws);
    _Float16* kws = (_Float16*)(ws + 131072);
    _Float16* w1f = (_Float16*)(ws + 262144);
    _Float16* w2f = (_Float16*)(ws + 262144 + 28672);

    proj_kernel<<<NB * NSEQ, 128, 0, stream>>>(x, Wd, bd, qws, kws);
    pack_w_kernel<<<(NFRAG + 255) / 256, 256, 0, stream>>>(W1, W2, w1f, w2f);
    pair_mlp_kernel<<<NB * NSEQ, 256, 0, stream>>>(
        qws, kws, (const v16h*)w1f, (const v16h*)w2f, b1, b2, out);
}
